// DCGNN_18614388261381
// MI455X (gfx1250) — compile-verified, matrix operand order checked by reading, not run on hardware
//
#include <hip/hip_runtime.h>
#include <math.h>

typedef __attribute__((ext_vector_type(16))) __bf16 v16bf;
typedef __attribute__((ext_vector_type(8)))  __bf16 bf16x8;
typedef __attribute__((ext_vector_type(8)))  float  v8f;

#define NS    1024
#define TSEQ  32
#define FEAT  16
#define FPAD  32
#define HID   192
#define GG    768      // 4*HID
#define DNF   384      // 2*HID
#define DU    128
#define NOUTC 2
#define MT    16       // stocks per LSTM block
#define LAMB  0.38f

// ---------------------------------------------------------------- helpers
__device__ __forceinline__ float sigf(float x) { return 1.f / (1.f + __expf(-x)); }

// ---------------------------------------------------------------- dynamic network
__global__ void k_prep(const float* __restrict__ x, const float* __restrict__ theta,
                       int* __restrict__ pid, int* __restrict__ offv) {
  int n = blockIdx.x * blockDim.x + threadIdx.x;
  if (n >= NS) return;
  float th = theta[n];
  float x0 = x[(n * TSEQ + (TSEQ - 2)) * FEAT + 0];
  float x1 = x[(n * TSEQ + (TSEQ - 1)) * FEAT + 0];
  int p0 = x0 > th ? 1 : (x0 < -th ? -1 : 0);
  int p1 = x1 > th ? 1 : (x1 < -th ? -1 : 0);
  pid[n]  = (p0 + 1) * 3 + (p1 + 1);
  offv[n] = (p1 + 1) * 3;     // Py uses yr = x[:,-1,0] with same theta
}

__global__ void k_adj_beta(const float* __restrict__ all_PP, const int* __restrict__ pid,
                           const int* __restrict__ offv, float* __restrict__ adj,
                           float* __restrict__ braw) {
  int i = blockIdx.x * blockDim.x + threadIdx.x;   // i = x*NS + y
  if (i >= NS * NS) return;
  int xi = i >> 10, yi = i & (NS - 1);
  const float* p = all_PP + (((size_t)i * 9 + pid[xi]) * 9 + offv[yi]);
  float a0 = p[0], a1 = p[1], a2 = p[2];
  adj[(size_t)i * 3 + 0] = a0;
  adj[(size_t)i * 3 + 1] = a1;
  adj[(size_t)i * 3 + 2] = a2;
  float nrm = sqrtf(a0 * a0 + a1 * a1 + a2 * a2);
  braw[i] = nrm > LAMB ? nrm : 0.f;
}

__global__ __launch_bounds__(256)
void k_softmax_col(const float* __restrict__ braw, float* __restrict__ Beta,
                   __bf16* __restrict__ BetaT) {
  int y = blockIdx.x, tid = threadIdx.x;
  __shared__ float red[256];
  float mx = -1e30f;
  for (int xx = tid; xx < NS; xx += 256) mx = fmaxf(mx, braw[(size_t)xx * NS + y]);
  red[tid] = mx; __syncthreads();
  for (int s = 128; s > 0; s >>= 1) { if (tid < s) red[tid] = fmaxf(red[tid], red[tid + s]); __syncthreads(); }
  mx = red[0]; __syncthreads();
  float sm = 0.f;
  for (int xx = tid; xx < NS; xx += 256) sm += __expf(braw[(size_t)xx * NS + y] - mx);
  red[tid] = sm; __syncthreads();
  for (int s = 128; s > 0; s >>= 1) { if (tid < s) red[tid] += red[tid + s]; __syncthreads(); }
  float inv = 1.f / red[0];
  for (int xx = tid; xx < NS; xx += 256) {
    float v = __expf(braw[(size_t)xx * NS + y] - mx) * inv;
    Beta[(size_t)xx * NS + y] = v;
    BetaT[(size_t)y * NS + xx] = (__bf16)v;   // row y of Beta^T
  }
}

// ---------------------------------------------------------------- conversions
__global__ void k_cvt_pad_x(const float* __restrict__ x, __bf16* __restrict__ xp) {
  int i = blockIdx.x * blockDim.x + threadIdx.x;   // over NS*TSEQ*FPAD
  if (i >= NS * TSEQ * FPAD) return;
  int row = i / FPAD, k = i - row * FPAD;
  xp[i] = (k < FEAT) ? (__bf16)x[row * FEAT + k] : (__bf16)0.f;
}

// gate-permute + pad + f32->bf16 for (2, GG, Kin) weights -> (2, GG, Kpad)
__global__ void k_conv_w(const float* __restrict__ src, __bf16* __restrict__ dst,
                         int Kin, int Kpad) {
  int i = blockIdx.x * blockDim.x + threadIdx.x;
  if (i >= 2 * GG * Kpad) return;
  int d = i / (GG * Kpad), r = i - d * GG * Kpad;
  int ng = r / Kpad, k = r - ng * Kpad;
  int hid = ng >> 2, gate = ng & 3;
  int og = gate * HID + hid;
  dst[i] = (k < Kin) ? (__bf16)src[((size_t)d * GG + og) * Kin + k] : (__bf16)0.f;
}

__global__ void k_conv_bias(const float* __restrict__ bih, const float* __restrict__ bhh,
                            float* __restrict__ bsum) {
  int i = blockIdx.x * blockDim.x + threadIdx.x;
  if (i >= 2 * GG) return;
  int d = i / GG, ng = i - d * GG;
  int og = (ng & 3) * HID + (ng >> 2);
  bsum[i] = bih[d * GG + og] + bhh[d * GG + og];
}

__global__ void k_transpose_cvt(const float* __restrict__ src, __bf16* __restrict__ dst,
                                int rows, int cols) {
  int i = blockIdx.x * blockDim.x + threadIdx.x;
  if (i >= rows * cols) return;
  int r = i / cols, c = i - r * cols;
  dst[(size_t)c * rows + r] = (__bf16)src[(size_t)r * cols + c];
}

__global__ void k_cvt(const float* __restrict__ src, __bf16* __restrict__ dst, int n) {
  int i = blockIdx.x * blockDim.x + threadIdx.x;
  if (i < n) dst[i] = (__bf16)src[i];
}

// ---------------------------------------------------------------- generic bf16 WMMA GEMM
// C[M,N] = A[M,K] @ B[N,K]^T (+ bias[N]);  A,B bf16 row-major, C f32.
// One wave -> one 16x16 tile; block = 8 waves stacked along M.
__global__ __launch_bounds__(256)
void k_gemm_bf16(const __bf16* __restrict__ A, const __bf16* __restrict__ B,
                 const float* __restrict__ bias, float* __restrict__ C,
                 int M, int Nn, int K) {
  const int lane = threadIdx.x & 31;
  const int wave = threadIdx.x >> 5;
  const int n0 = blockIdx.x * 16;
  const int m0 = (blockIdx.y * 8 + wave) * 16;
  if (m0 >= M || n0 >= Nn) return;
  const int row   = lane & 15;
  const int khalf = (lane >> 4) * 8;
  const __bf16* Ar = A + (size_t)(m0 + row) * K;
  const __bf16* Br = B + (size_t)(n0 + row) * K;
  v8f acc = {};
  for (int k0 = 0; k0 < K; k0 += 32) {
    if (k0 + 32 < K) {          // hide HBM/L2 latency behind the current WMMA
      __builtin_prefetch(Ar + k0 + 32, 0, 1);
      __builtin_prefetch(Br + k0 + 32, 0, 1);
    }
    v16bf a, b;
#pragma unroll
    for (int e = 0; e < 8; ++e) {
      a[e]     = Ar[k0 + khalf + e];
      a[e + 8] = Ar[k0 + 16 + khalf + e];
      b[e]     = Br[k0 + khalf + e];
      b[e + 8] = Br[k0 + 16 + khalf + e];
    }
    acc = __builtin_amdgcn_wmma_f32_16x16x32_bf16(false, a, false, b, (short)0, acc,
                                                  false, false);
  }
  const int cn = n0 + (lane & 15);
  float ba = bias ? bias[cn] : 0.f;
#pragma unroll
  for (int r = 0; r < 8; ++r) {
    int cm = m0 + r + (lane >> 4) * 8;
    C[(size_t)cm * Nn + cn] = acc[r] + ba;
  }
}

// ---------------------------------------------------------------- LSTM recurrence
// Whh (GG x HID, gate-permuted bf16) pinned in LDS (288 KB), staged by
// GLOBAL_LOAD_ASYNC_TO_LDS_B128 (ASYNCcnt path — no VGPR round-trip).
// 16 stocks / block.  Per step: C := gx[t] (bias folded); C += h @ Whh^T via
// 288 WMMAs; gates in VALU with cell state in registers; h written to LDS
// (A-fragment source) and to the global bf16 sequence buffer.
__global__ __launch_bounds__(256)
void k_lstm_rec(const float* __restrict__ gx,      // (NS*TSEQ, GG) f32, permuted cols
                const __bf16* __restrict__ Whh,    // (GG, HID) bf16, permuted rows
                __bf16* __restrict__ hout,         // (NS, TSEQ, ldh)
                int ldh, int col_off,
                float* __restrict__ last_out,      // (NS, DNF) or null
                int backward) {
  __shared__ __bf16 sW[GG * HID];    // 294912 B
  __shared__ __bf16 sG[MT * GG];     //  24576 B
  __shared__ __bf16 sH[MT * HID];    //   6144 B
  const int tid = threadIdx.x;
  const int lane = tid & 31, wave = tid >> 5;
  const int stock0 = blockIdx.x * MT;

  // Async DMA Whh -> LDS.  For LDS-aperture flat addresses, addr[31:0] is the
  // LDS byte address (ISA 10.2), so truncating the generic pointer gives the
  // VDST value the async op needs, while normal sW accesses stay on ds_*.
  {
    const unsigned lds0 = (unsigned)(size_t)sW;
    const char* g0 = (const char*)Whh;
    for (int i = tid; i < (GG * HID * 2) / 16; i += 256) {
      unsigned laddr = lds0 + (unsigned)i * 16u;
      unsigned long long gaddr = (unsigned long long)(size_t)(g0 + (size_t)i * 16);
      asm volatile("global_load_async_to_lds_b128 %0, %1, off"
                   :: "v"(laddr), "v"(gaddr) : "memory");
    }
  }
  for (int i = tid; i < MT * HID; i += 256) sH[i] = (__bf16)0.f;
  asm volatile("s_wait_asynccnt 0x0" ::: "memory");
  __syncthreads();

  float c[12];
#pragma unroll
  for (int j = 0; j < 12; ++j) c[j] = 0.f;

  const int row = lane & 15;
  const int khalf = (lane >> 4) * 8;

  for (int s = 0; s < TSEQ; ++s) {
    const int t = backward ? (TSEQ - 1 - s) : s;
    v8f acc[6];
    // C init from gx (bias folded)
#pragma unroll
    for (int j2 = 0; j2 < 6; ++j2) {
      int cn = (wave * 6 + j2) * 16 + (lane & 15);
#pragma unroll
      for (int r = 0; r < 8; ++r) {
        int cm = r + (lane >> 4) * 8;
        acc[j2][r] = gx[((size_t)(stock0 + cm) * TSEQ + t) * GG + cn];
      }
    }
    // h @ Whh^T
    for (int k0 = 0; k0 < HID; k0 += 32) {
      v16bf a;
#pragma unroll
      for (int e = 0; e < 8; ++e) {
        a[e]     = sH[row * HID + k0 + khalf + e];
        a[e + 8] = sH[row * HID + k0 + 16 + khalf + e];
      }
#pragma unroll
      for (int j2 = 0; j2 < 6; ++j2) {
        int nc = (wave * 6 + j2) * 16 + (lane & 15);
        v16bf b;
#pragma unroll
        for (int e = 0; e < 8; ++e) {
          b[e]     = sW[nc * HID + k0 + khalf + e];
          b[e + 8] = sW[nc * HID + k0 + 16 + khalf + e];
        }
        acc[j2] = __builtin_amdgcn_wmma_f32_16x16x32_bf16(false, a, false, b, (short)0,
                                                          acc[j2], false, false);
      }
    }
    // spill pre-activations to LDS for the elementwise gate pass
#pragma unroll
    for (int j2 = 0; j2 < 6; ++j2) {
      int cn = (wave * 6 + j2) * 16 + (lane & 15);
#pragma unroll
      for (int r = 0; r < 8; ++r) {
        int cm = r + (lane >> 4) * 8;
        sG[cm * GG + cn] = (__bf16)acc[j2][r];
      }
    }
    __syncthreads();
    // gates: thread owns 12 (m,hid) pairs; gate columns interleaved (i,f,g,o)
#pragma unroll
    for (int j = 0; j < 12; ++j) {
      int q = tid * 12 + j;
      int m = q / HID, hid = q - m * HID;
      float gi = (float)sG[m * GG + hid * 4 + 0];
      float gf = (float)sG[m * GG + hid * 4 + 1];
      float gc = (float)sG[m * GG + hid * 4 + 2];
      float go = (float)sG[m * GG + hid * 4 + 3];
      float cc = sigf(gf) * c[j] + sigf(gi) * tanhf(gc);
      c[j] = cc;
      float hh = sigf(go) * tanhf(cc);
      __bf16 hb = (__bf16)hh;
      sH[m * HID + hid] = hb;
      hout[((size_t)(stock0 + m) * TSEQ + t) * ldh + col_off + hid] = hb;
      if (last_out && t == TSEQ - 1)
        last_out[(size_t)(stock0 + m) * DNF + col_off + hid] = hh;
    }
    __syncthreads();
  }
}

// ---------------------------------------------------------------- einsum term + bias
// acc[y,:] += (sum_x Beta[x,y]*adj[x,y,e]) @ Wa(3,DU) + b(DU)
__global__ __launch_bounds__(256)
void k_extra_agg(const float* __restrict__ Beta, const float* __restrict__ adj,
                 const float* __restrict__ Wa, const float* __restrict__ bagg,
                 float* __restrict__ acc) {
  int y = blockIdx.x, tid = threadIdx.x;
  __shared__ float r0[256], r1[256], r2[256];
  float s0 = 0.f, s1 = 0.f, s2 = 0.f;
  for (int xx = tid; xx < NS; xx += 256) {
    float b = Beta[(size_t)xx * NS + y];
    const float* ap = adj + ((size_t)xx * NS + y) * 3;
    s0 += b * ap[0]; s1 += b * ap[1]; s2 += b * ap[2];
  }
  r0[tid] = s0; r1[tid] = s1; r2[tid] = s2; __syncthreads();
  for (int s = 128; s > 0; s >>= 1) {
    if (tid < s) { r0[tid] += r0[tid + s]; r1[tid] += r1[tid + s]; r2[tid] += r2[tid + s]; }
    __syncthreads();
  }
  if (tid < DU) {
    float o = r0[0] * Wa[tid] + r1[0] * Wa[DU + tid] + r2[0] * Wa[2 * DU + tid] + bagg[tid];
    acc[(size_t)y * DU + tid] += o;
  }
}

// ---------------------------------------------------------------- head
__global__ void k_head(const float* __restrict__ nfeat, const float* __restrict__ n1,
                       const float* __restrict__ n2, const float* __restrict__ Wout,
                       const float* __restrict__ bout, float* __restrict__ out) {
  int n = blockIdx.x * blockDim.x + threadIdx.x;
  if (n >= NS) return;
  float a0 = bout[0], a1 = bout[1];
  for (int e = 0; e < DNF; ++e) {
    float v = nfeat[(size_t)n * DNF + e];
    a0 += v * Wout[e * 2]; a1 += v * Wout[e * 2 + 1];
  }
  for (int e = 0; e < DU; ++e) {
    float v = n1[(size_t)n * DU + e];
    a0 += v * Wout[(DNF + e) * 2]; a1 += v * Wout[(DNF + e) * 2 + 1];
  }
  for (int e = 0; e < DU; ++e) {
    float v = n2[(size_t)n * DU + e];
    a0 += v * Wout[(DNF + DU + e) * 2]; a1 += v * Wout[(DNF + DU + e) * 2 + 1];
  }
  out[n * 2 + 0] = tanhf(a0);
  out[n * 2 + 1] = tanhf(a1);
}

// ---------------------------------------------------------------- host launcher
static constexpr size_t alup(size_t x) { return (x + 255) & ~size_t(255); }

extern "C" void kernel_launch(void* const* d_in, const int* in_sizes, int n_in,
                              void* d_out, int out_size, void* d_ws, size_t ws_size,
                              hipStream_t stream) {
  const float* input  = (const float*)d_in[0];
  const float* theta  = (const float*)d_in[1];
  const float* all_PP = (const float*)d_in[2];
  const float* Wih0   = (const float*)d_in[3];
  const float* Whh0   = (const float*)d_in[4];
  const float* bih0   = (const float*)d_in[5];
  const float* bhh0   = (const float*)d_in[6];
  const float* Wih1   = (const float*)d_in[7];
  const float* Whh1   = (const float*)d_in[8];
  const float* bih1   = (const float*)d_in[9];
  const float* bhh1   = (const float*)d_in[10];
  const float* Wagg1  = (const float*)d_in[11];
  const float* bagg1  = (const float*)d_in[12];
  const float* Wagg2  = (const float*)d_in[13];
  const float* bagg2  = (const float*)d_in[14];
  const float* Wout   = (const float*)d_in[15];
  const float* bout   = (const float*)d_in[16];
  float* out = (float*)d_out;

  char* ws = (char*)d_ws;
  size_t o = 0;
  auto carve = [&](size_t bytes) { size_t r = o; o = alup(o + bytes); return r; };
  int*    pid     = (int*)   (ws + carve(NS * 4));
  int*    offv    = (int*)   (ws + carve(NS * 4));
  float*  adj     = (float*) (ws + carve((size_t)NS * NS * 3 * 4));
  float*  braw    = (float*) (ws + carve((size_t)NS * NS * 4));
  float*  Beta    = (float*) (ws + carve((size_t)NS * NS * 4));
  __bf16* BetaT   = (__bf16*)(ws + carve((size_t)NS * NS * 2));
  __bf16* xpad    = (__bf16*)(ws + carve((size_t)NS * TSEQ * FPAD * 2));
  __bf16* wih0b   = (__bf16*)(ws + carve((size_t)2 * GG * FPAD * 2));
  __bf16* whh0b   = (__bf16*)(ws + carve((size_t)2 * GG * HID * 2));
  __bf16* wih1b   = (__bf16*)(ws + carve((size_t)2 * GG * DNF * 2));
  __bf16* whh1b   = (__bf16*)(ws + carve((size_t)2 * GG * HID * 2));
  float*  bs0     = (float*) (ws + carve((size_t)2 * GG * 4));
  float*  bs1     = (float*) (ws + carve((size_t)2 * GG * 4));
  float*  gx      = (float*) (ws + carve((size_t)NS * TSEQ * GG * 4));
  __bf16* h1b     = (__bf16*)(ws + carve((size_t)NS * TSEQ * DNF * 2));
  __bf16* h2b     = (__bf16*)(ws + carve((size_t)NS * TSEQ * DNF * 2));
  float*  nlast   = (float*) (ws + carve((size_t)NS * DNF * 4));
  __bf16* nlastb  = (__bf16*)(ws + carve((size_t)NS * DNF * 2));
  __bf16* wfT     = (__bf16*)(ws + carve((size_t)DU * DNF * 2));
  __bf16* wf2T    = (__bf16*)(ws + carve((size_t)DU * DU * 2));
  float*  P       = (float*) (ws + carve((size_t)NS * DU * 4));
  __bf16* PT      = (__bf16*)(ws + carve((size_t)DU * NS * 2));
  float*  n1      = (float*) (ws + carve((size_t)NS * DU * 4));
  __bf16* n1b     = (__bf16*)(ws + carve((size_t)NS * DU * 2));
  float*  Q       = (float*) (ws + carve((size_t)NS * DU * 4));
  __bf16* QT      = (__bf16*)(ws + carve((size_t)DU * NS * 2));
  float*  n2      = (float*) (ws + carve((size_t)NS * DU * 4));
  (void)ws_size; (void)n_in; (void)in_sizes; (void)out_size;

  // ---- dynamic network
  k_prep<<<NS / 256, 256, 0, stream>>>(input, theta, pid, offv);
  k_adj_beta<<<(NS * NS) / 256, 256, 0, stream>>>(all_PP, pid, offv, adj, braw);
  k_softmax_col<<<NS, 256, 0, stream>>>(braw, Beta, BetaT);

  // ---- conversions
  k_cvt_pad_x<<<(NS * TSEQ * FPAD) / 256, 256, 0, stream>>>(input, xpad);
  k_conv_w<<<(2 * GG * FPAD) / 256, 256, 0, stream>>>(Wih0, wih0b, FEAT, FPAD);
  k_conv_w<<<(2 * GG * HID) / 256, 256, 0, stream>>>(Whh0, whh0b, HID, HID);
  k_conv_w<<<(2 * GG * DNF) / 256, 256, 0, stream>>>(Wih1, wih1b, DNF, DNF);
  k_conv_w<<<(2 * GG * HID) / 256, 256, 0, stream>>>(Whh1, whh1b, HID, HID);
  k_conv_bias<<<(2 * GG) / 256, 256, 0, stream>>>(bih0, bhh0, bs0);
  k_conv_bias<<<(2 * GG) / 256, 256, 0, stream>>>(bih1, bhh1, bs1);
  k_transpose_cvt<<<(DNF * DU) / 256, 256, 0, stream>>>(Wagg1, wfT, DNF, DU);
  k_transpose_cvt<<<(DU * DU) / 256, 256, 0, stream>>>(Wagg2, wf2T, DU, DU);

  const int MSEQ = NS * TSEQ;  // 32768
  dim3 blk(256);
  // ---- BiLSTM layer 0
  k_gemm_bf16<<<dim3(GG / 16, MSEQ / 128), blk, 0, stream>>>(xpad, wih0b, bs0, gx, MSEQ, GG, FPAD);
  k_lstm_rec<<<NS / MT, blk, 0, stream>>>(gx, whh0b, h1b, DNF, 0, nullptr, 0);
  k_gemm_bf16<<<dim3(GG / 16, MSEQ / 128), blk, 0, stream>>>(xpad, wih0b + (size_t)GG * FPAD, bs0 + GG, gx, MSEQ, GG, FPAD);
  k_lstm_rec<<<NS / MT, blk, 0, stream>>>(gx, whh0b + (size_t)GG * HID, h1b, DNF, HID, nullptr, 1);
  // ---- BiLSTM layer 1
  k_gemm_bf16<<<dim3(GG / 16, MSEQ / 128), blk, 0, stream>>>(h1b, wih1b, bs1, gx, MSEQ, GG, DNF);
  k_lstm_rec<<<NS / MT, blk, 0, stream>>>(gx, whh1b, h2b, DNF, 0, nlast, 0);
  k_gemm_bf16<<<dim3(GG / 16, MSEQ / 128), blk, 0, stream>>>(h1b, wih1b + (size_t)GG * DNF, bs1 + GG, gx, MSEQ, GG, DNF);
  k_lstm_rec<<<NS / MT, blk, 0, stream>>>(gx, whh1b + (size_t)GG * HID, h2b, DNF, HID, nlast, 1);

  // ---- causal aggregation 1:  n1 = Beta^T @ (nlast @ Wf) + einsum + b
  k_cvt<<<(NS * DNF) / 256, 256, 0, stream>>>(nlast, nlastb, NS * DNF);
  k_gemm_bf16<<<dim3(DU / 16, NS / 128), blk, 0, stream>>>(nlastb, wfT, nullptr, P, NS, DU, DNF);
  k_transpose_cvt<<<(NS * DU) / 256, 256, 0, stream>>>(P, PT, NS, DU);
  k_gemm_bf16<<<dim3(DU / 16, NS / 128), blk, 0, stream>>>(BetaT, PT, nullptr, n1, NS, DU, NS);
  k_extra_agg<<<NS, 256, 0, stream>>>(Beta, adj, Wagg1 + (size_t)DNF * DU, bagg1, n1);
  // ---- causal aggregation 2
  k_cvt<<<(NS * DU) / 256, 256, 0, stream>>>(n1, n1b, NS * DU);
  k_gemm_bf16<<<dim3(DU / 16, NS / 128), blk, 0, stream>>>(n1b, wf2T, nullptr, Q, NS, DU, DU);
  k_transpose_cvt<<<(NS * DU) / 256, 256, 0, stream>>>(Q, QT, NS, DU);
  k_gemm_bf16<<<dim3(DU / 16, NS / 128), blk, 0, stream>>>(BetaT, QT, nullptr, n2, NS, DU, NS);
  k_extra_agg<<<NS, 256, 0, stream>>>(Beta, adj, Wagg2 + (size_t)DU * DU, bagg2, n2);

  // ---- output head
  k_head<<<NS / 256, 256, 0, stream>>>(nlast, n1, n2, Wout, bout, out);
}
